// LongRangeLayer_86191403696596
// MI455X (gfx1250) — matrix operand order repaired and checked by reading, not compiled, not run on hardware
//
#include <hip/hip_runtime.h>

// CDNA5 / gfx1250: wave32, WMMA 16x16x32 bf16 -> f32
typedef __attribute__((ext_vector_type(16))) __bf16 v16bf;
typedef __attribute__((ext_vector_type(8)))  __bf16 v8bf;   // 16 bytes
typedef __attribute__((ext_vector_type(8)))  float  v8f;
typedef __attribute__((ext_vector_type(4)))  float  v4f;

#define IMG_H 64
#define IMG_W 64
#define NCH   128
#define NGRP  16
#define PSTR  88   // LDS patch row stride in bf16 (176B, 16B-aligned rows, conflict-free)

// --------------------------------------------------------------------------
// 17x17 depthwise conv (replicate pad 8) via Toeplitz-filter WMMA GEMM.
// out[x] = sum_t w[t]*in[x+t]  ==>  A[16x32] Toeplitz(w_row) x B[32x16] input.
// One wave per (b, g, jj, tileRow): loads a 32x80 padded bf16 patch to LDS,
// then 4 output tiles of 16x16, each = 17 accumulated v_wmma_f32_16x16x32_bf16.
// --------------------------------------------------------------------------
__global__ __launch_bounds__(32)
void lr_conv_wmma(const float* __restrict__ srcA, const float* __restrict__ srcB,
                  const float* __restrict__ filt, float* __restrict__ dst,
                  int nJ, int jA0, int jB0, int jF0, int jD0)
{
    __shared__ __bf16 patch[32 * PSTR];
    __shared__ float  wf[17 * 17];

    const int lane = threadIdx.x;          // 0..31, one wave per block
    int idx = blockIdx.x;
    const int r  = idx & 3;   idx >>= 2;   // tile row (y0 = 16*r)
    const int jj = idx % nJ;  idx /= nJ;
    const int g  = idx % NGRP; idx /= NGRP;
    const int b  = idx;

    const int cA = g * 8 + jA0 + jj;
    const int cB = g * 8 + jB0 + jj;
    const int cD = g * 8 + jD0 + jj;
    const float* fsrc = filt + (size_t)(jF0 + jj) * 289;
    const float* pa = srcA + (size_t)(b * NCH + cA) * (IMG_H * IMG_W);
    const float* pb = srcB + (size_t)(b * NCH + cB) * (IMG_H * IMG_W);
    float*       pd = dst  + (size_t)(b * NCH + cD) * (IMG_H * IMG_W);

    // filter row cache
    for (int i = lane; i < 289; i += 32) wf[i] = fsrc[i];

    const int y0 = r * 16;
    // padded patch: rows y0..y0+31, cols 0..79 of the 80x80 edge-padded image
    for (int t = lane; t < 32 * 80; t += 32) {
        const int row = t / 80, col = t % 80;
        int iy = y0 + row - 8; iy = iy < 0 ? 0 : (iy > 63 ? 63 : iy);
        int ix = col - 8;      ix = ix < 0 ? 0 : (ix > 63 ? 63 : ix);
        const float v = pa[iy * IMG_W + ix] - pb[iy * IMG_W + ix];
        patch[row * PSTR + col] = (__bf16)(v > 0.f ? v : 0.f);
    }
    __syncthreads();

    // Build the 17 Toeplitz A fragments (one per filter row), contiguous k order, same as B:
    // lane L: m = L&15; k = (L>>4)*16 + 2v+e
    const int m    = lane & 15;
    const int half = lane >> 4;
    v16bf afrag[17];
#pragma unroll
    for (int ry = 0; ry < 17; ++ry) {
        const float* wr = wf + ry * 17;
        v16bf a;
#pragma unroll
        for (int v = 0; v < 8; ++v) {
#pragma unroll
            for (int e = 0; e < 2; ++e) {
                const int k = half * 16 + 2 * v + e;
                const int d = k - m;
                const float val = (d >= 0 && d <= 16) ? wr[d] : 0.f;
                a[2 * v + e] = (__bf16)val;
            }
        }
        afrag[ry] = a;
    }

    const int n = lane & 15;   // B/D column = output row within tile
#pragma unroll
    for (int tx = 0; tx < 4; ++tx) {
        const int x0 = tx * 16;
        v8f acc = {};
#pragma unroll
        for (int ry = 0; ry < 17; ++ry) {
            // B layout: lane L holds K = (L>>4)*16 + (0..15), N = L&15.
            // 16 consecutive bf16, 16B-aligned -> two ds_load_b128.
            const __bf16* src = &patch[(ry + n) * PSTR + x0 + half * 16];
            v16bf bfr;
            *((v8bf*)&bfr)     = *((const v8bf*)src);
            *((v8bf*)&bfr + 1) = *((const v8bf*)(src + 8));
            acc = __builtin_amdgcn_wmma_f32_16x16x32_bf16(
                false, afrag[ry], false, bfr, (short)0, acc, false, false);
        }
        // D layout: lane L, vgpr v -> out(y = y0 + (L&15), x = x0 + 8*(L>>4) + v).
        // 8 consecutive floats per lane, 16B aligned -> two global_store_b128.
        float* orow = &pd[(y0 + n) * IMG_W + x0 + 8 * half];
        v4f lo = { acc[0], acc[1], acc[2], acc[3] };
        v4f hi = { acc[4], acc[5], acc[6], acc[7] };
        *((v4f*)orow)       = lo;
        *((v4f*)(orow + 4)) = hi;
    }
}

// --------------------------------------------------------------------------
// Fused: circular 5-tap orientation gaussian -> 7x7 separable reflect blur
// (in LDS) -> final elementwise combine. One block per (b, c) image.
// --------------------------------------------------------------------------
__global__ __launch_bounds__(256)
void mix_blur_combine(const float* __restrict__ x, const float* __restrict__ netp,
                      float* __restrict__ out)
{
    __shared__ float t0[64 * 65];
    __shared__ float t1[64 * 65];

    const int tid = threadIdx.x;
    const int b = blockIdx.x >> 7;
    const int c = blockIdx.x & 127;
    const int s  = c & 7;
    const int cg = c & ~7;

    float g5[5], g7[7];
    {
        float sum = 0.f;
        for (int i = 0; i < 5; ++i) { g5[i] = __expf(-(float)((i - 2) * (i - 2)) / (2.f * 0.25f)); sum += g5[i]; }
        for (int i = 0; i < 5; ++i) g5[i] /= sum;
    }
    {
        float sum = 0.f;
        for (int i = 0; i < 7; ++i) { g7[i] = __expf(-(float)((i - 3) * (i - 3)) / (2.f * 64.f)); sum += g7[i]; }
        for (int i = 0; i < 7; ++i) g7[i] /= sum;
    }

    const size_t base = (size_t)b * NCH * 4096;
    float np[16];

#pragma unroll
    for (int i = 0; i < 16; ++i) {
        const int p = tid + i * 256;
        const int y = p >> 6, xx = p & 63;
        float acc = 0.f;
#pragma unroll
        for (int k = 0; k < 5; ++k) {
            const int ch = cg + ((s + k - 2) & 7);
            const float v = netp[base + (size_t)ch * 4096 + p];
            acc += g5[k] * v;
            if (k == 2) np[i] = v;
        }
        t0[y * 65 + xx] = acc;
    }
    __syncthreads();

    // horizontal 7-tap, reflect padding
#pragma unroll
    for (int i = 0; i < 16; ++i) {
        const int p = tid + i * 256;
        const int y = p >> 6, xx = p & 63;
        float acc = 0.f;
#pragma unroll
        for (int k = 0; k < 7; ++k) {
            int xr = xx + k - 3;
            xr = xr < 0 ? -xr : (xr > 63 ? 126 - xr : xr);
            acc += g7[k] * t0[y * 65 + xr];
        }
        t1[y * 65 + xx] = acc;
    }
    __syncthreads();

    // vertical 7-tap + final combine
    const float* xc = x   + base + (size_t)c * 4096;
    float*       oc = out + base + (size_t)c * 4096;
#pragma unroll
    for (int i = 0; i < 16; ++i) {
        const int p = tid + i * 256;
        const int y = p >> 6, xx = p & 63;
        float acc = 0.f;
#pragma unroll
        for (int k = 0; k < 7; ++k) {
            int yr = y + k - 3;
            yr = yr < 0 ? -yr : (yr > 63 ? 126 - yr : yr);
            acc += g7[k] * t1[yr * 65 + xx];
        }
        const float xv = xc[p];
        oc[p] = 0.001f * xv * (1.f + 5.f * np[i]) / (0.2f + 2.f * acc);
    }
}

extern "C" void kernel_launch(void* const* d_in, const int* in_sizes, int n_in,
                              void* d_out, int out_size, void* d_ws, size_t ws_size,
                              hipStream_t stream)
{
    const float* x  = (const float*)d_in[0];   // [32,128,64,64] f32
    const float* lr = (const float*)d_in[1];   // [8,17,17] f32
    float* out  = (float*)d_out;
    float* netp = (float*)d_ws;                // 32*128*64*64 f32 = 64 MiB scratch

    const int B = 32;
    // Stage A: j=0..3, v = relu(x[j] - x[j+2])            -> netp[0..3]
    lr_conv_wmma<<<B * NGRP * 4 * 4, 32, 0, stream>>>(x, x,    lr, netp, 4, 0, 2, 0, 0);
    // Stage B: j=4,5, v = relu(x[j] - netp[j-2])          -> netp[4,5]
    lr_conv_wmma<<<B * NGRP * 2 * 4, 32, 0, stream>>>(x, netp, lr, netp, 2, 4, 2, 4, 4);
    // Stage C: j=6,7, v = relu(x[j] - netp[j-2])          -> netp[6,7]
    lr_conv_wmma<<<B * NGRP * 2 * 4, 32, 0, stream>>>(x, netp, lr, netp, 2, 6, 4, 6, 6);
    // Orientation mix + 7x7 blur + combine
    mix_blur_combine<<<B * NCH, 256, 0, stream>>>(x, netp, out);
}
